// TQEngine_5437428597383
// MI455X (gfx1250) — matrix operand order repaired and checked by reading, not compiled, not run on hardware
//
#include <hip/hip_runtime.h>

#define TQ_DIM 256

typedef unsigned int   u32;
typedef unsigned short u16;
typedef __attribute__((ext_vector_type(16))) __bf16 v16bf;
typedef __attribute__((ext_vector_type(8)))  float  v8f;

struct Frag  { uint4 lo, hi; };
union  FragU { Frag f; v16bf v; };

// round-to-nearest-even f32 -> bf16
static __device__ __forceinline__ u16 f2bf(float f){
  u32 u = __float_as_uint(f);
  u32 r = u + 0x7fffu + ((u >> 16) & 1u);
  return (u16)(r >> 16);
}
static __device__ __forceinline__ u32 packbf(float a, float b){
  return (u32)f2bf(a) | ((u32)f2bf(b) << 16);
}

// Stage half (128 B-rows) of a 256x256 f32 matrix into a 64KB LDS bf16 panel:
//   panel[nl*128 + kp] packs B[2kp][n], B[2kp+1][n],  n = h*128 + nl.
// transpose=1: B = M       (column gather of M)
// transpose=0: B = M^T     (row-major copy of M)
static __device__ __forceinline__ void stage_half(const float* __restrict__ M,
                                                  u32* panel, int t, int transpose, int h){
  if (transpose){
    for (int i = 0; i < 128; ++i){
      int gid = i*128 + t;
      int kp = gid >> 7, nl = gid & 127;          // consecutive t -> consecutive n (coalesced)
      int n  = h*128 + nl;
      float a = M[(2*kp)*TQ_DIM + n];
      float b = M[(2*kp+1)*TQ_DIM + n];
      panel[nl*128 + kp] = packbf(a, b);
    }
  } else {
    for (int i = 0; i < 128; ++i){
      int gid = i*128 + t;
      int nl = gid >> 7, kp = gid & 127;          // consecutive t -> consecutive kp (coalesced)
      float2 v = ((const float2*)M)[(h*128 + nl)*128 + kp];
      panel[nl*128 + kp] = packbf(v.x, v.y);
    }
  }
}

// Load the wave's 8 A fragments (16x32 bf16 each) once per GEMM phase.
// A layout: lane<16 holds K 0..7 / 16..23, lane>=16 holds K 8..15 / 24..31 (row = lr)
static __device__ __forceinline__ void load_afrags(FragU* afr, const u32* bufA,
                                                   int lr, int half){
  #pragma unroll
  for (int kt = 0; kt < 8; ++kt){
    afr[kt].f.lo = *(const uint4*)(bufA + lr*128 + kt*16 +     half*4);
    afr[kt].f.hi = *(const uint4*)(bufA + lr*128 + kt*16 + 8 + half*4);
  }
}

// One 16x16 output tile, K=256 in 8 steps of v_wmma_f32_16x16x32_bf16.
// B (32x16 bf16): lane<16 holds K 0..7 / 16..23, lane>=16 holds K 8..15 / 24..31, lane lr = column n.
// ntl is the tile index inside the current 128-column half-panel (0..7).
static __device__ __forceinline__ v8f gemm_tile(const FragU* afr, const u32* panel,
                                                int lr, int half, int ntl){
  v8f acc = {0.f,0.f,0.f,0.f,0.f,0.f,0.f,0.f};
  const int n = ntl*16 + lr;
  #pragma unroll
  for (int kt = 0; kt < 8; ++kt){
    FragU b;
    b.f.lo = *(const uint4*)(panel + n*128 + kt*16 +     half*4);
    b.f.hi = *(const uint4*)(panel + n*128 + kt*16 + 8 + half*4);
    acc = __builtin_amdgcn_wmma_f32_16x16x32_bf16(false, afr[kt].v, false, b.v,
                                                  (short)0, acc, false, false);
  }
  return acc;
}

// Scatter one bf16 C tile into the wave's row-major ping-pong buffer.
static __device__ __forceinline__ void dump_tile(u16* dst, int half, int lr,
                                                 int nt, uint4 kv){
  union { uint4 q; u32 p[4]; } u; u.q = kv;
  #pragma unroll
  for (int j = 0; j < 4; ++j){
    int M = 8*half + 2*j;
    dst[M*TQ_DIM     + nt*16 + lr] = (u16)(u.p[j] & 0xffffu);
    dst[(M+1)*TQ_DIM + nt*16 + lr] = (u16)(u.p[j] >> 16);
  }
}

extern "C" __global__ void __launch_bounds__(128)
tq_fused_kernel(const float* __restrict__ x,
                const float* __restrict__ Pi,
                const float* __restrict__ cen,
                const float* __restrict__ bnd,
                const float* __restrict__ S,
                float* __restrict__ out)
{
  // 64KB shared half-panel (bf16) + 4 waves x (2 x 8KB) ping-pong A buffers = 128KB
  __shared__ __align__(16) unsigned char smem[65536 + 4*16384];
  u32* panel = (u32*)smem;
  u32* bufU  = (u32*)(smem + 65536);
  u16* bufH  = (u16*)(smem + 65536);

  const int t    = threadIdx.x;
  const int w    = t >> 5;
  const int lane = t & 31;
  const int half = lane >> 4;
  const int lr   = lane & 15;
  u32* buf0  = bufU + w*4096;   // ping: 16 rows x 128 bf16-pairs
  u32* buf1  = buf0 + 2048;     // pong
  u16* buf0H = bufH + w*8192;
  u16* buf1H = buf0H + 4096;
  const int rowBase = blockIdx.x*64 + w*16;

  const float b0 = bnd[0], b1 = bnd[1], b2 = bnd[2];
  const float c0 = cen[0], c1 = cen[1], c2 = cen[2], c3 = cen[3];

  // ---- Stage A: row norms + unit rows (bf16) into ping buffer ----
  const float* xrow = x + (rowBase + lr)*TQ_DIM + half*128;
  float sumsq = 0.f;
  #pragma unroll 8
  for (int c = 0; c < 32; ++c){
    float4 v = ((const float4*)xrow)[c];
    sumsq += v.x*v.x + v.y*v.y + v.z*v.z + v.w*v.w;
  }
  float nrm = sumsq + __shfl_xor(sumsq, 16, 32);
  nrm = sqrtf(nrm);
  const float inv = 1.f / (nrm + 1e-10f);
  #pragma unroll 8
  for (int c = 0; c < 32; ++c){
    float4 v = ((const float4*)xrow)[c];
    buf0[lr*128 + half*64 + 2*c    ] = packbf(v.x*inv, v.y*inv);
    buf0[lr*128 + half*64 + 2*c + 1] = packbf(v.z*inv, v.w*inv);
  }
  // broadcast row norms to the lanes that will scale x_mse tiles (C-layout rows)
  float nm[8];
  #pragma unroll
  for (int r = 0; r < 8; ++r) nm[r] = __shfl(nrm, 8*half + r, 32);

  FragU afr[8];

  // ---- GEMM1: y = unit @ Pi, fused 2-bit quantize -> centroid; y_hat -> pong ----
  load_afrags(afr, buf0, lr, half);
  #pragma unroll 1
  for (int h = 0; h < 2; ++h){
    __syncthreads();                      // prior panel readers done
    stage_half(Pi, panel, t, 1, h);
    __syncthreads();                      // panel ready
    #pragma unroll
    for (int ntl = 0; ntl < 8; ++ntl){
      v8f y = gemm_tile(afr, panel, lr, half, ntl);
      u32 p[4];
      #pragma unroll
      for (int j = 0; j < 4; ++j){
        float y0 = y[2*j], y1 = y[2*j+1];
        // searchsorted(boundaries, y) -> centroid, branchless monotone nest
        float h0 = (y0 > b1) ? ((y0 > b2) ? c3 : c2) : ((y0 > b0) ? c1 : c0);
        float h1 = (y1 > b1) ? ((y1 > b2) ? c3 : c2) : ((y1 > b0) ? c1 : c0);
        p[j] = packbf(h0, h1);
      }
      dump_tile(buf1H, half, lr, h*8 + ntl, make_uint4(p[0], p[1], p[2], p[3]));
    }
  }

  // ---- GEMM2: x_mse = (y_hat @ Pi^T)*norm; residual -> ping; res-norm partials ----
  float rn[8] = {0.f,0.f,0.f,0.f,0.f,0.f,0.f,0.f};
  load_afrags(afr, buf1, lr, half);
  #pragma unroll 1
  for (int h = 0; h < 2; ++h){
    __syncthreads();
    stage_half(Pi, panel, t, 0, h);       // B = Pi^T
    __syncthreads();
    #pragma unroll
    for (int ntl = 0; ntl < 8; ++ntl){
      v8f a = gemm_tile(afr, panel, lr, half, ntl);
      const int nt  = h*8 + ntl;
      const int col = nt*16 + lr;
      u32 p[4];
      #pragma unroll
      for (int j = 0; j < 4; ++j){
        int M0 = 8*half + 2*j;
        int i0 = (rowBase + M0)*TQ_DIM + col;
        int i1 = i0 + TQ_DIM;
        float xm0 = a[2*j]   * nm[2*j];
        float xm1 = a[2*j+1] * nm[2*j+1];
        out[i0] = xm0; out[i1] = xm1;     // x_mse parked in out
        float r0 = x[i0] - xm0;
        float r1 = x[i1] - xm1;
        rn[2*j] += r0*r0; rn[2*j+1] += r1*r1;
        p[j] = packbf(r0, r1);
      }
      dump_tile(buf0H, half, lr, nt, make_uint4(p[0], p[1], p[2], p[3]));
    }
  }
  #pragma unroll
  for (int m = 1; m < 16; m <<= 1){
    #pragma unroll
    for (int r = 0; r < 8; ++r) rn[r] += __shfl_xor(rn[r], m, 32);
  }
  float rsc[8];                           // QJL_SCALE * ||residual||
  #pragma unroll
  for (int r = 0; r < 8; ++r) rsc[r] = 0.004895758348888673f * sqrtf(rn[r]);

  // ---- GEMM3: projected = residual @ S^T, fused sign hash; signs -> pong ----
  load_afrags(afr, buf0, lr, half);
  #pragma unroll 1
  for (int h = 0; h < 2; ++h){
    __syncthreads();
    stage_half(S, panel, t, 0, h);        // B = S^T
    __syncthreads();
    #pragma unroll
    for (int ntl = 0; ntl < 8; ++ntl){
      v8f pr = gemm_tile(afr, panel, lr, half, ntl);
      u32 p[4];
      #pragma unroll
      for (int j = 0; j < 4; ++j){
        float s0 = pr[2*j]   > 0.f ? 1.f : -1.f;
        float s1 = pr[2*j+1] > 0.f ? 1.f : -1.f;
        p[j] = packbf(s0, s1);
      }
      dump_tile(buf1H, half, lr, h*8 + ntl, make_uint4(p[0], p[1], p[2], p[3]));
    }
  }

  // ---- GEMM4: x_qjl = (signs @ S) * rsc;  out = x_mse + x_qjl ----
  load_afrags(afr, buf1, lr, half);
  #pragma unroll 1
  for (int h = 0; h < 2; ++h){
    __syncthreads();
    stage_half(S, panel, t, 1, h);        // B = S
    __syncthreads();
    #pragma unroll
    for (int ntl = 0; ntl < 8; ++ntl){
      v8f q = gemm_tile(afr, panel, lr, half, ntl);
      const int col = (h*8 + ntl)*16 + lr;
      #pragma unroll
      for (int j = 0; j < 4; ++j){
        int M0 = 8*half + 2*j;
        int i0 = (rowBase + M0)*TQ_DIM + col;
        int i1 = i0 + TQ_DIM;
        out[i0] += rsc[2*j]   * q[2*j];   // same-thread store->load RMW: in order
        out[i1] += rsc[2*j+1] * q[2*j+1];
      }
    }
  }
}

extern "C" void kernel_launch(void* const* d_in, const int* in_sizes, int n_in,
                              void* d_out, int out_size, void* d_ws, size_t ws_size,
                              hipStream_t stream) {
  (void)n_in; (void)out_size; (void)d_ws; (void)ws_size;
  const float* x   = (const float*)d_in[0];
  const float* Pi  = (const float*)d_in[1];
  const float* cen = (const float*)d_in[2];
  const float* bnd = (const float*)d_in[3];
  const float* S   = (const float*)d_in[4];
  float* out = (float*)d_out;
  const int N      = in_sizes[0] / TQ_DIM;   // 131072 rows
  const int blocks = N / 64;                 // 64 rows per workgroup (4 waves x 16)
  tq_fused_kernel<<<blocks, 128, 0, stream>>>(x, Pi, cen, bnd, S, out);
}
